// GraphNetwork_67877663146533
// MI455X (gfx1250) — matrix operand order repaired and checked by reading, not compiled, not run on hardware
//
#include <hip/hip_runtime.h>

typedef __attribute__((ext_vector_type(16))) __bf16 v16bf;
typedef __attribute__((ext_vector_type(8)))  __bf16 v8bf;
typedef __attribute__((ext_vector_type(8)))  float  v8f;

#define BB     32
#define NN     400
#define J2C    5
#define FF     64
#define DIN0   3
#define NLAY   6
#define MPAD   416      // 13 * 32
#define KSTEPS 13
#define MTILES 25       // 400 / 16

// ---------------------------------------------------------------------------
// WW (B,N,N,J2) f32  ->  WWb (B,J2,N,MPAD) bf16, zero padded in m
// ---------------------------------------------------------------------------
__global__ void k_conv_ww(const float* __restrict__ ww, __bf16* __restrict__ wwb) {
    long idx = (long)blockIdx.x * blockDim.x + threadIdx.x;
    const long total = (long)BB * J2C * NN * MPAD;
    if (idx >= total) return;
    int m = (int)(idx % MPAD);
    long t = idx / MPAD;
    int n = (int)(t % NN); t /= NN;
    int j = (int)(t % J2C);
    int b = (int)(t / J2C);
    float v = 0.f;
    if (m < NN) v = ww[(((long)b * NN + n) * NN + m) * J2C + j];
    wwb[idx] = (__bf16)v;
}

// ---------------------------------------------------------------------------
// X (B,N,DIN) f32 -> XbT (B,F,MPAD) bf16 transposed+zero padded (f>=DIN, m>=N)
// ---------------------------------------------------------------------------
__global__ void k_conv_x0(const float* __restrict__ x, __bf16* __restrict__ xbt) {
    int idx = blockIdx.x * blockDim.x + threadIdx.x;
    const int total = BB * FF * MPAD;
    if (idx >= total) return;
    int m = idx % MPAD;
    int t = idx / MPAD;
    int f = t % FF;
    int b = t / FF;
    float v = 0.f;
    if (m < NN && f < DIN0) v = x[((long)b * NN + m) * DIN0 + f];
    xbt[idx] = (__bf16)v;
}

// ---------------------------------------------------------------------------
// Fused gmul (bf16 WMMA) + linear (+ReLU on first half) per 16-row tile.
// grid = B*25 blocks, 128 threads (4 waves).  wave w = output col-tile w.
// mode 0: gconv layer -> write O (B*N,F) + per-block BN partials
// mode 1: last layer  -> L2-normalize rows, write bf16 embeddings eb (B,N,F)
// ---------------------------------------------------------------------------
__global__ void __launch_bounds__(128) k_gemm(
    const __bf16* __restrict__ wwb,     // (B,J2,N,MPAD)
    const __bf16* __restrict__ xbt,     // (B,F,MPAD)
    const float* __restrict__ w1, const float* __restrict__ b1,
    const float* __restrict__ w2, const float* __restrict__ b2,
    float* __restrict__ obuf,           // (B*N,F)      mode 0
    float* __restrict__ partials,       // (nblk,2*F)   mode 0
    __bf16* __restrict__ eb,            // (B,N,F)      mode 1
    int dcur, int nin, int mode)
{
    __shared__ float hl[16][J2C * FF + 1];   // gmul result tile (16 x 320)
    __shared__ float ol[16][FF];             // linear output tile (16 x 64)
    __shared__ float rn[16];

    const int blk  = blockIdx.x;
    const int b    = blk / MTILES;
    const int mt   = blk % MTILES;
    const int lane = threadIdx.x & 31;
    const int ct   = threadIdx.x >> 5;       // col tile 0..3
    const int r16  = lane & 15;
    const int hi   = lane >> 4;

    v8f acc[J2C];
    for (int j = 0; j < J2C; ++j)
        for (int i = 0; i < 8; ++i) acc[j][i] = 0.f;

    // A: row = lane&15 of this row-tile;  B: col = lane&15 of this col-tile
    const __bf16* aBase = wwb + ((long)b * J2C * NN + (mt * 16 + r16)) * MPAD;
    const __bf16* bBase = xbt + ((long)b * FF + ct * 16 + r16) * MPAD + hi * 16;

    for (int ks = 0; ks < KSTEPS; ++ks) {
        // B fragment: 16 contiguous bf16 (K half selected by lane>>4)
        v16bf bf = *(const v16bf*)(bBase + ks * 32);
        for (int j = 0; j < J2C; ++j) {
            const __bf16* ap = aBase + (long)j * NN * MPAD + ks * 32;
            v8bf lo = *(const v8bf*)(ap + hi * 16);        // K {0..7} / {16..23}
            v8bf hg = *(const v8bf*)(ap + hi * 16 + 8);    // K {8..15} / {24..31}
            v16bf af;
            for (int i = 0; i < 8; ++i) { af[i] = lo[i]; af[8 + i] = hg[i]; }
            acc[j] = __builtin_amdgcn_wmma_f32_16x16x32_bf16(
                false, af, false, bf, (short)0, acc[j], false, false);
        }
    }

    // C/D layout: element r of v8f -> M = r + 8*(lane>=16), N = lane&15
    for (int j = 0; j < J2C; ++j)
        for (int r = 0; r < 8; ++r)
            hl[r + hi * 8][j * FF + ct * 16 + r16] = acc[j][r];
    __syncthreads();

    // fused linear: 16 rows x 64 outputs, 8 per thread
    for (int t = threadIdx.x; t < 16 * FF; t += 128) {
        int rr = t >> 6;
        int f  = t & 63;
        const float* wrow;
        float s;
        if (f < 32) { wrow = w1 + (long)f * nin;        s = b1[f]; }
        else        { wrow = w2 + (long)(f - 32) * nin; s = b2[f - 32]; }
        if (dcur == FF) {
            for (int c = 0; c < nin; ++c) s += wrow[c] * hl[rr][c];
        } else {
            for (int c = 0; c < nin; ++c) {
                int j = c / dcur, dd = c - j * dcur;
                s += wrow[c] * hl[rr][j * FF + dd];
            }
        }
        if (mode == 0 && f < 32) s = fmaxf(s, 0.f);   // ReLU on h1 path
        ol[rr][f] = s;
    }
    __syncthreads();

    if (mode == 0) {
        for (int t = threadIdx.x; t < 16 * FF; t += 128) {
            int rr = t >> 6, f = t & 63;
            obuf[(long)(b * NN + mt * 16 + rr) * FF + f] = ol[rr][f];
        }
        if (threadIdx.x < FF) {
            int f = threadIdx.x;
            float s = 0.f, ss = 0.f;
            for (int rr = 0; rr < 16; ++rr) { float v = ol[rr][f]; s += v; ss += v * v; }
            partials[(long)blk * 2 * FF + f]      = s;
            partials[(long)blk * 2 * FF + FF + f] = ss;
        }
    } else {
        if (threadIdx.x < 16) {
            float s = 0.f;
            for (int f = 0; f < FF; ++f) { float v = ol[threadIdx.x][f]; s += v * v; }
            rn[threadIdx.x] = rsqrtf(s);
        }
        __syncthreads();
        for (int t = threadIdx.x; t < 16 * FF; t += 128) {
            int rr = t >> 6, f = t & 63;
            eb[(long)(b * NN + mt * 16 + rr) * FF + f] = (__bf16)(ol[rr][f] * rn[rr]);
        }
    }
}

// ---------------------------------------------------------------------------
// Reduce per-block partials -> mu[F], rstd[F]  (biased variance)
// ---------------------------------------------------------------------------
__global__ void k_bnstat(const float* __restrict__ partials, int nblk,
                         float* __restrict__ stats) {
    int f = threadIdx.x;
    if (f >= FF) return;
    float s = 0.f, ss = 0.f;
    for (int i = 0; i < nblk; ++i) {
        s  += partials[(long)i * 2 * FF + f];
        ss += partials[(long)i * 2 * FF + FF + f];
    }
    const float inv = 1.f / (float)(BB * NN);
    float mu  = s * inv;
    float var = ss * inv - mu * mu;
    stats[f]      = mu;
    stats[FF + f] = rsqrtf(var + 1e-5f);
}

// ---------------------------------------------------------------------------
// BatchNorm (batch stats) + InstanceNorm over F, write next activations
// directly as transposed padded bf16 (B,F,MPAD).  grid = B*N, block = 64.
// ---------------------------------------------------------------------------
__global__ void __launch_bounds__(64) k_bnin(
    const float* __restrict__ obuf, const float* __restrict__ stats,
    const float* __restrict__ gamma, const float* __restrict__ beta,
    __bf16* __restrict__ xbt)
{
    __shared__ float s1[FF], s2[FF];
    int rowg = blockIdx.x;
    int f = threadIdx.x;
    int b = rowg / NN, n = rowg % NN;
    float v = obuf[(long)rowg * FF + f];
    v = (v - stats[f]) * stats[FF + f] * gamma[f] + beta[f];
    s1[f] = v; s2[f] = v * v;
    __syncthreads();
    for (int off = 32; off > 0; off >>= 1) {
        if (f < off) { s1[f] += s1[f + off]; s2[f] += s2[f + off]; }
        __syncthreads();
    }
    float mu  = s1[0] / FF;
    float var = s2[0] / FF - mu * mu;
    float out = (v - mu) * rsqrtf(var + 1e-5f);
    xbt[((long)b * FF + f) * MPAD + n] = (__bf16)out;
}

// ---------------------------------------------------------------------------
// Gram: out[b,n,m] = sum_f e[b,n,f]*e[b,m,f] - 1000*(n==m).
// 1 wave per 16x16 output tile, K = 64 = 2 WMMA steps.
// ---------------------------------------------------------------------------
__global__ void __launch_bounds__(32) k_gram(const __bf16* __restrict__ eb,
                                             float* __restrict__ out) {
    int blk = blockIdx.x;
    int nt  = blk % MTILES; int t = blk / MTILES;
    int mt  = t % MTILES;   int b = t / MTILES;
    int lane = threadIdx.x;
    int r16 = lane & 15, hi = lane >> 4;

    v8f acc;
    for (int i = 0; i < 8; ++i) acc[i] = 0.f;

    const __bf16* aBase = eb + ((long)b * NN + mt * 16 + r16) * FF;
    const __bf16* bBase = eb + ((long)b * NN + nt * 16 + r16) * FF + hi * 16;
    for (int ks = 0; ks < 2; ++ks) {
        v16bf bf = *(const v16bf*)(bBase + ks * 32);
        v8bf lo = *(const v8bf*)(aBase + ks * 32 + hi * 16);
        v8bf hg = *(const v8bf*)(aBase + ks * 32 + hi * 16 + 8);
        v16bf af;
        for (int i = 0; i < 8; ++i) { af[i] = lo[i]; af[8 + i] = hg[i]; }
        acc = __builtin_amdgcn_wmma_f32_16x16x32_bf16(
            false, af, false, bf, (short)0, acc, false, false);
    }
    for (int r = 0; r < 8; ++r) {
        int row  = r + hi * 8;
        int grow = mt * 16 + row;
        int gcol = nt * 16 + r16;
        float val = acc[r];
        if (grow == gcol) val -= 1000.f;
        out[((long)b * NN + grow) * NN + gcol] = val;
    }
}

// ---------------------------------------------------------------------------
extern "C" void kernel_launch(void* const* d_in, const int* in_sizes, int n_in,
                              void* d_out, int out_size, void* d_ws, size_t ws_size,
                              hipStream_t stream) {
    const float* WW = (const float*)d_in[0];
    const float* X  = (const float*)d_in[1];

    char* p = (char*)d_ws;
    __bf16* wwb = (__bf16*)p; p += (size_t)BB * J2C * NN * MPAD * 2;   // 53.25 MB
    __bf16* xbt = (__bf16*)p; p += (size_t)BB * FF * MPAD * 2;         // 1.70 MB
    float*  obuf = (float*)p; p += (size_t)BB * NN * FF * 4;           // 3.28 MB
    float*  parts = (float*)p; p += (size_t)BB * MTILES * 2 * FF * 4;  // 0.41 MB
    float*  stats = (float*)p; p += 2 * FF * 4;
    __bf16* ebuf = (__bf16*)p; p += (size_t)BB * NN * FF * 2;          // 1.64 MB

    {   // WW -> bf16 (one HBM pass; bf16 copy then lives in 192MB L2)
        long total = (long)BB * J2C * NN * MPAD;
        k_conv_ww<<<(int)((total + 255) / 256), 256, 0, stream>>>(WW, wwb);
    }
    {   // X -> transposed padded bf16
        int total = BB * FF * MPAD;
        k_conv_x0<<<(total + 255) / 256, 256, 0, stream>>>(X, xbt);
    }

    int dcur = DIN0;
    for (int i = 0; i < NLAY + 1; ++i) {
        const float* w1 = (const float*)d_in[2 + 6 * i + 0];
        const float* b1 = (const float*)d_in[2 + 6 * i + 1];
        const float* w2 = (const float*)d_in[2 + 6 * i + 2];
        const float* b2 = (const float*)d_in[2 + 6 * i + 3];
        const float* ga = (const float*)d_in[2 + 6 * i + 4];
        const float* be = (const float*)d_in[2 + 6 * i + 5];
        int nin = J2C * dcur;
        k_gemm<<<BB * MTILES, 128, 0, stream>>>(wwb, xbt, w1, b1, w2, b2,
                                                obuf, parts, nullptr, dcur, nin, 0);
        k_bnstat<<<1, 64, 0, stream>>>(parts, BB * MTILES, stats);
        k_bnin<<<BB * NN, 64, 0, stream>>>(obuf, stats, ga, be, xbt);
        dcur = FF;
    }

    // last layer: linear(320->64), L2-normalize rows, emit bf16 embeddings
    const float* wl = (const float*)d_in[2 + 6 * (NLAY + 1)];
    const float* bl = (const float*)d_in[2 + 6 * (NLAY + 1) + 1];
    k_gemm<<<BB * MTILES, 128, 0, stream>>>(wwb, xbt,
                                            wl, bl, wl + 32 * (J2C * FF), bl + 32,
                                            obuf, nullptr, ebuf, FF, J2C * FF, 1);

    // gram + diag
    k_gram<<<BB * MTILES * MTILES, 32, 0, stream>>>(ebuf, (float*)d_out);
}
